// MoEDispatch_53068615909641
// MI455X (gfx1250) — matrix operand order repaired and checked by reading, not compiled, not run on hardware
//
#include <hip/hip_runtime.h>
#include <math.h>

// ---------------------------------------------------------------------------
// MoE forward for MI455X (gfx1250, wave32, WMMA).
// x:[4,2048,1024] f32, W_router:[1024,16], W1:[16,1024,4096], b1:[16,4096],
// W2:[16,4096,1024], b2:[16,1024].  N=8192 tokens, capacity=640, top-2.
// Expert GEMMs run on v_wmma_f32_16x16x32_bf16 (fp32->bf16 convert in LDS
// staging; fp32 accumulation). 172 GFLOP vs 536MB weights => HBM-bound on
// the bf16 matrix path (~23us of HBM traffic at 23.3 TB/s).
// ---------------------------------------------------------------------------

#define NTOK 8192
#define DMODEL 1024
#define DHID 4096
#define NEXP 16
#define CAP 640

typedef __attribute__((ext_vector_type(8)))  float  v8f;
typedef __attribute__((ext_vector_type(16))) __bf16 v16bf;
typedef __attribute__((ext_vector_type(8)))  __bf16 v8bf;

union BFrag { v16bf v; v8bf h[2]; };
struct alignas(8) bf16x4 { __bf16 a, b, c, d; };

// ---------------------------------------------------------------------------
// 1. Router: logits = x @ W_router, softmax over 16 experts. One wave/token.
// ---------------------------------------------------------------------------
__global__ __launch_bounds__(128) void router_softmax_kernel(
    const float* __restrict__ x, const float* __restrict__ Wr,
    float* __restrict__ probs)
{
  const int wave = threadIdx.x >> 5, lane = threadIdx.x & 31;
  const int t = blockIdx.x * 4 + wave;
  const float* xt = x + (size_t)t * DMODEL;
  float acc[NEXP];
#pragma unroll
  for (int e = 0; e < NEXP; ++e) acc[e] = 0.f;
  for (int k = lane; k < DMODEL; k += 32) {
    const float xv = xt[k];
    const float4* wr = (const float4*)&Wr[k * NEXP];
    const float4 w0 = wr[0], w1 = wr[1], w2 = wr[2], w3 = wr[3];
    acc[0]  += xv * w0.x; acc[1]  += xv * w0.y; acc[2]  += xv * w0.z; acc[3]  += xv * w0.w;
    acc[4]  += xv * w1.x; acc[5]  += xv * w1.y; acc[6]  += xv * w1.z; acc[7]  += xv * w1.w;
    acc[8]  += xv * w2.x; acc[9]  += xv * w2.y; acc[10] += xv * w2.z; acc[11] += xv * w2.w;
    acc[12] += xv * w3.x; acc[13] += xv * w3.y; acc[14] += xv * w3.z; acc[15] += xv * w3.w;
  }
#pragma unroll
  for (int e = 0; e < NEXP; ++e)
#pragma unroll
    for (int off = 16; off > 0; off >>= 1)
      acc[e] += __shfl_xor(acc[e], off, 32);
  if (lane == 0) {
    float mx = acc[0];
#pragma unroll
    for (int e = 1; e < NEXP; ++e) mx = fmaxf(mx, acc[e]);
    float ex[NEXP], s = 0.f;
#pragma unroll
    for (int e = 0; e < NEXP; ++e) { ex[e] = expf(acc[e] - mx); s += ex[e]; }
    const float inv = 1.f / s;
#pragma unroll
    for (int e = 0; e < NEXP; ++e) probs[(size_t)t * NEXP + e] = ex[e] * inv;
  }
}

// ---------------------------------------------------------------------------
// 2. Sinkhorn step A: row-normalize in place + per-block column partial sums
//    (deterministic: fixed-order in-block reduction, no float atomics).
// ---------------------------------------------------------------------------
__global__ __launch_bounds__(256) void sinkhorn_rownorm_kernel(
    float* __restrict__ probs, float* __restrict__ partial)
{
  __shared__ float sm[256][NEXP + 1];
  const int t = blockIdx.x * 256 + threadIdx.x;
  float4* row = (float4*)&probs[(size_t)t * NEXP];
  float4 r0 = row[0], r1 = row[1], r2 = row[2], r3 = row[3];
  const float s = r0.x + r0.y + r0.z + r0.w + r1.x + r1.y + r1.z + r1.w +
                  r2.x + r2.y + r2.z + r2.w + r3.x + r3.y + r3.z + r3.w;
  const float inv = 1.f / s;
  float v[NEXP] = {r0.x*inv, r0.y*inv, r0.z*inv, r0.w*inv, r1.x*inv, r1.y*inv,
                   r1.z*inv, r1.w*inv, r2.x*inv, r2.y*inv, r2.z*inv, r2.w*inv,
                   r3.x*inv, r3.y*inv, r3.z*inv, r3.w*inv};
  float4* rv = (float4*)v;
  row[0] = rv[0]; row[1] = rv[1]; row[2] = rv[2]; row[3] = rv[3];
#pragma unroll
  for (int e = 0; e < NEXP; ++e) sm[threadIdx.x][e] = v[e];
  __syncthreads();
  if (threadIdx.x < NEXP) {
    float cs = 0.f;
    for (int j = 0; j < 256; ++j) cs += sm[j][threadIdx.x];
    partial[blockIdx.x * NEXP + threadIdx.x] = cs;
  }
}

__global__ void reduce_colsum_kernel(const float* __restrict__ partial,
                                     float* __restrict__ colsum, int nblk)
{
  const int e = threadIdx.x;
  if (e < NEXP) {
    float s = 0.f;
    for (int b = 0; b < nblk; ++b) s += partial[b * NEXP + e];
    colsum[e] = s;
  }
}

// Sinkhorn step B: probs[:,e] *= (N/E) / colsum[e]
__global__ __launch_bounds__(256) void sinkhorn_colscale_kernel(
    float* __restrict__ probs, const float* __restrict__ colsum)
{
  __shared__ float sc[NEXP];
  if (threadIdx.x < NEXP)
    sc[threadIdx.x] = ((float)NTOK / (float)NEXP) / colsum[threadIdx.x];
  __syncthreads();
  const int t = blockIdx.x * 256 + threadIdx.x;
  float* row = &probs[(size_t)t * NEXP];
#pragma unroll
  for (int e = 0; e < NEXP; ++e) row[e] *= sc[e];
}

// ---------------------------------------------------------------------------
// 3. Top-2 per token + normalized weights + final column partial sums.
// ---------------------------------------------------------------------------
__global__ __launch_bounds__(256) void topk_kernel(
    const float* __restrict__ probs, int* __restrict__ idx,
    float* __restrict__ wn, float* __restrict__ partial)
{
  __shared__ float sm[256][NEXP + 1];
  const int t = blockIdx.x * 256 + threadIdx.x;
  const float* row = &probs[(size_t)t * NEXP];
  float v[NEXP];
#pragma unroll
  for (int e = 0; e < NEXP; ++e) { v[e] = row[e]; sm[threadIdx.x][e] = v[e]; }
  int i0 = 0; float v0 = v[0];
#pragma unroll
  for (int e = 1; e < NEXP; ++e) if (v[e] > v0) { v0 = v[e]; i0 = e; }
  int i1 = (i0 == 0) ? 1 : 0; float v1 = v[i1];
#pragma unroll
  for (int e = 0; e < NEXP; ++e)
    if (e != i0 && v[e] > v1) { v1 = v[e]; i1 = e; }
  const float inv = 1.f / (v0 + v1);
  idx[2 * t] = i0; idx[2 * t + 1] = i1;
  wn[2 * t] = v0 * inv; wn[2 * t + 1] = v1 * inv;
  __syncthreads();
  if (threadIdx.x < NEXP) {
    float cs = 0.f;
    for (int j = 0; j < 256; ++j) cs += sm[j][threadIdx.x];
    partial[blockIdx.x * NEXP + threadIdx.x] = cs;
  }
}

// ---------------------------------------------------------------------------
// 4. Ordered slot allocation: exact sequential (t,k)-order positions via
//    wave same-expert masks + hierarchical per-expert prefix. One block.
// ---------------------------------------------------------------------------
__global__ __launch_bounds__(1024) void dispatch_scan_kernel(
    const int* __restrict__ flat_e, const float* __restrict__ wn,
    int* __restrict__ safe_pos, float* __restrict__ weff,
    int* __restrict__ slot_token, int* __restrict__ token_counts)
{
  __shared__ int wave_cnt[32][NEXP];
  __shared__ int wave_off[32][NEXP];
  __shared__ int base[NEXP];
  __shared__ int chunk_tot[NEXP];
  const int tid = threadIdx.x, lane = tid & 31, wave = tid >> 5;
  if (tid < NEXP) base[tid] = 0;
  __syncthreads();
  for (int c = 0; c < NTOK * 2; c += 1024) {
    const int i = c + tid;
    const int e = flat_e[i];
    unsigned same = 0u;
#pragma unroll
    for (int j = 0; j < 32; ++j) {
      const int ej = __shfl(e, j, 32);
      same |= (unsigned)(ej == e) << j;
    }
    const unsigned lt = (1u << lane) - 1u;
    const int prefix = __popc(same & lt);
    const int total = __popc(same);
    if (tid < 32 * NEXP) ((int*)wave_cnt)[tid] = 0;
    __syncthreads();
    if (prefix == 0) wave_cnt[wave][e] = total;   // unique (wave,e) writer
    __syncthreads();
    if (tid < NEXP) {
      int run = 0;
      for (int w = 0; w < 32; ++w) { wave_off[w][tid] = run; run += wave_cnt[w][tid]; }
      chunk_tot[tid] = run;
    }
    __syncthreads();
    const int pos = base[e] + wave_off[wave][e] + prefix;
    const int ok = pos < CAP;
    safe_pos[i] = ok ? pos : 0;
    weff[i] = ok ? wn[i] : 0.f;
    if (ok) slot_token[e * CAP + pos] = i >> 1;
    __syncthreads();
    if (tid < NEXP) base[tid] += chunk_tot[tid];
    __syncthreads();
  }
  if (tid < NEXP) token_counts[tid] = base[tid] < CAP ? base[tid] : CAP;
}

// ---------------------------------------------------------------------------
// 5. Gather tokens into [E, CAP, D] expert buffers (zero empty slots).
// ---------------------------------------------------------------------------
__global__ __launch_bounds__(128) void gather_kernel(
    const float* __restrict__ x, const int* __restrict__ slot_token,
    float* __restrict__ ein)
{
  const int s = blockIdx.x;
  const int t = slot_token[s];
  float4* dst = (float4*)&ein[(size_t)s * DMODEL];
  if (t >= 0) {
    const float4* src = (const float4*)&x[(size_t)t * DMODEL];
    for (int j = threadIdx.x; j < DMODEL / 4; j += 128) dst[j] = src[j];
  } else {
    const float4 z = {0.f, 0.f, 0.f, 0.f};
    for (int j = threadIdx.x; j < DMODEL / 4; j += 128) dst[j] = z;
  }
}

// ---------------------------------------------------------------------------
// 6. Per-expert GEMM on WMMA bf16 tensor path.
//    C[e] = act(A[e](MxK) @ W[e](KxN) + bias[e]).
//    Block tile 128x128, 8 waves of 32x64 (2x4 tiles of 16x16), BK=64.
//    fp32 -> bf16 conversion during LDS staging; B staged n-major so each
//    lane's 16 K values (two runs of 8) are two ds_load_b128s.
// ---------------------------------------------------------------------------
#define BM 128
#define BN 128
#define BK 64
#define LDK 72   // padded bf16 row stride: 144B, keeps 16B alignment

__global__ __launch_bounds__(256) void expert_gemm_bf16_kernel(
    const float* __restrict__ A, const float* __restrict__ W,
    const float* __restrict__ bias, float* __restrict__ C,
    int M, int N, int K, int apply_gelu)
{
  const int e = blockIdx.z;
  const int n0 = blockIdx.x * BN;
  const int m0 = blockIdx.y * BM;
  const float* Ae = A + (size_t)e * M * K;
  const float* We = W + (size_t)e * K * N;
  const float* be = bias + (size_t)e * N;
  float* Ce = C + (size_t)e * M * N;

  __shared__ __align__(16) __bf16 As[BM][LDK];
  __shared__ __align__(16) __bf16 Bt[BN][LDK];   // [n][k]

  const int tid = threadIdx.x;
  const int lane = tid & 31;
  const int wave = tid >> 5;
  const int wm = wave & 3;     // 4 waves along M, 32 rows each
  const int wn = wave >> 2;    // 2 waves along N, 64 cols each
  const int lm = lane & 15;
  const int hi = lane >> 4;    // half-wave select per WMMA VGPR layout

  v8f acc[2][4];
#pragma unroll
  for (int mi = 0; mi < 2; ++mi)
#pragma unroll
    for (int ni = 0; ni < 4; ++ni) acc[mi][ni] = (v8f)0.0f;

  for (int k0 = 0; k0 < K; k0 += BK) {
    if (k0 + BK < K) {  // prefetch next K-stage (global_prefetch_b8)
      __builtin_prefetch(&Ae[(size_t)(m0 + (tid >> 1)) * K + k0 + BK + (tid & 1) * 32], 0, 3);
      __builtin_prefetch(&We[(size_t)(k0 + BK + (tid >> 5)) * N + n0 + (tid & 31) * 4], 0, 3);
    }
    {   // stage A: 128x64 f32 -> bf16, row-major
      const int r0 = tid >> 4;
      const int c4 = (tid & 15) * 4;
      for (int r = r0; r < BM; r += 16) {
        const float4 v = *(const float4*)&Ae[(size_t)(m0 + r) * K + k0 + c4];
        bf16x4 t = {(__bf16)v.x, (__bf16)v.y, (__bf16)v.z, (__bf16)v.w};
        *(bf16x4*)&As[r][c4] = t;   // ds_store_b64
      }
    }
    {   // stage B: 64x128 f32 -> bf16, transposed to [n][k]
      const int kr0 = tid >> 5;
      const int c4 = (tid & 31) * 4;
      for (int k = kr0; k < BK; k += 8) {
        const float4 v = *(const float4*)&We[(size_t)(k0 + k) * N + n0 + c4];
        Bt[c4 + 0][k] = (__bf16)v.x;
        Bt[c4 + 1][k] = (__bf16)v.y;
        Bt[c4 + 2][k] = (__bf16)v.z;
        Bt[c4 + 3][k] = (__bf16)v.w;
      }
    }
    __syncthreads();
#pragma unroll
    for (int ks = 0; ks < BK; ks += 32) {
      BFrag af[2], bf[4];
#pragma unroll
      for (int mi = 0; mi < 2; ++mi) {      // A 16x32: K {0..7|8..15}, {16..23|24..31}
        const int row = wm * 32 + mi * 16 + lm;
        af[mi].h[0] = *(const v8bf*)&As[row][ks + hi * 8];
        af[mi].h[1] = *(const v8bf*)&As[row][ks + 16 + hi * 8];
      }
#pragma unroll
      for (int ni = 0; ni < 4; ++ni) {      // B 32x16: K {0..7|8..15}, {16..23|24..31}
        const int col = wn * 64 + ni * 16 + lm;
        bf[ni].h[0] = *(const v8bf*)&Bt[col][ks + hi * 8];
        bf[ni].h[1] = *(const v8bf*)&Bt[col][ks + 16 + hi * 8];
      }
#pragma unroll
      for (int mi = 0; mi < 2; ++mi)
#pragma unroll
        for (int ni = 0; ni < 4; ++ni)
          acc[mi][ni] = __builtin_amdgcn_wmma_f32_16x16x32_bf16(
              false, af[mi].v, false, bf[ni].v, (short)0, acc[mi][ni],
              false, false);
    }
    __syncthreads();
  }

  // epilogue: bias (+ exact GELU) and store. C layout: VGPR r -> row r (lanes
  // 0-15) / row 8+r (lanes 16-31), col = lane%16.
#pragma unroll
  for (int mi = 0; mi < 2; ++mi) {
#pragma unroll
    for (int ni = 0; ni < 4; ++ni) {
      const int row0 = m0 + wm * 32 + mi * 16 + hi * 8;
      const int col = n0 + wn * 64 + ni * 16 + lm;
      const float bv = be[col];
#pragma unroll
      for (int r = 0; r < 8; ++r) {
        float o = acc[mi][ni][r] + bv;
        if (apply_gelu) o = 0.5f * o * (1.0f + erff(o * 0.70710678118654752f));
        Ce[(size_t)(row0 + r) * N + col] = o;
      }
    }
  }
}

// ---------------------------------------------------------------------------
// 7. Combine: y[t] = w0*out[e0,p0] + w1*out[e1,p1]   (w==0 for dropped)
// ---------------------------------------------------------------------------
__global__ __launch_bounds__(256) void combine_kernel(
    const float* __restrict__ eout, const int* __restrict__ idx,
    const int* __restrict__ pos, const float* __restrict__ w,
    float* __restrict__ y)
{
  const int t = blockIdx.x;
  const int e0 = idx[2 * t], e1 = idx[2 * t + 1];
  const int p0 = pos[2 * t], p1 = pos[2 * t + 1];
  const float w0 = w[2 * t], w1 = w[2 * t + 1];
  const float4* o0 = (const float4*)&eout[((size_t)e0 * CAP + p0) * DMODEL];
  const float4* o1 = (const float4*)&eout[((size_t)e1 * CAP + p1) * DMODEL];
  float4* yt = (float4*)&y[(size_t)t * DMODEL];
  const int j = threadIdx.x;     // 256 threads * float4 = 1024
  const float4 a = o0[j], b = o1[j];
  float4 r;
  r.x = w0 * a.x + w1 * b.x; r.y = w0 * a.y + w1 * b.y;
  r.z = w0 * a.z + w1 * b.z; r.w = w0 * a.w + w1 * b.w;
  yt[j] = r;
}

__global__ void aux_loss_kernel(const float* __restrict__ colsum_final,
                                const int* __restrict__ token_counts,
                                float* __restrict__ out)
{
  if (threadIdx.x == 0) {
    float s = 0.f;
    for (int e = 0; e < NEXP; ++e)
      s += (colsum_final[e] / (float)NTOK) *
           ((float)token_counts[e] / (float)NTOK);
    out[0] = 0.01f * (float)NEXP * s;
  }
}

// ---------------------------------------------------------------------------
extern "C" void kernel_launch(void* const* d_in, const int* in_sizes, int n_in,
                              void* d_out, int out_size, void* d_ws, size_t ws_size,
                              hipStream_t stream)
{
  const float* x  = (const float*)d_in[0];
  const float* Wr = (const float*)d_in[1];
  const float* W1 = (const float*)d_in[2];
  const float* b1 = (const float*)d_in[3];
  const float* W2 = (const float*)d_in[4];
  const float* b2 = (const float*)d_in[5];
  float* y = (float*)d_out;

  char* p = (char*)d_ws;
  auto alloc = [&](size_t bytes) {
    char* r = p; p += (bytes + 255) & ~(size_t)255; return r;
  };
  float* probs    = (float*)alloc((size_t)NTOK * NEXP * 4);   // 512 KB
  float* partial  = (float*)alloc((size_t)32 * NEXP * 4);
  float* colsum   = (float*)alloc(NEXP * 4);
  int*   tk_idx   = (int*)  alloc((size_t)NTOK * 2 * 4);
  float* tk_w     = (float*)alloc((size_t)NTOK * 2 * 4);
  int*   d_pos    = (int*)  alloc((size_t)NTOK * 2 * 4);
  float* d_w      = (float*)alloc((size_t)NTOK * 2 * 4);
  int*   slot_tok = (int*)  alloc((size_t)NEXP * CAP * 4);
  int*   tok_cnt  = (int*)  alloc(NEXP * 4);
  float* ein      = (float*)alloc((size_t)NEXP * CAP * DMODEL * 4);  // 40 MB
  float* hbuf     = (float*)alloc((size_t)NEXP * CAP * DHID * 4);    // 160 MB
  float* eout     = (float*)alloc((size_t)NEXP * CAP * DMODEL * 4);  // 40 MB

  hipMemsetAsync(slot_tok, 0xFF, (size_t)NEXP * CAP * 4, stream);  // -1

  router_softmax_kernel<<<NTOK / 4, 128, 0, stream>>>(x, Wr, probs);

  for (int it = 0; it < 3; ++it) {
    sinkhorn_rownorm_kernel<<<NTOK / 256, 256, 0, stream>>>(probs, partial);
    reduce_colsum_kernel<<<1, NEXP, 0, stream>>>(partial, colsum, NTOK / 256);
    sinkhorn_colscale_kernel<<<NTOK / 256, 256, 0, stream>>>(probs, colsum);
  }

  topk_kernel<<<NTOK / 256, 256, 0, stream>>>(probs, tk_idx, tk_w, partial);
  reduce_colsum_kernel<<<1, NEXP, 0, stream>>>(partial, colsum, NTOK / 256);

  dispatch_scan_kernel<<<1, 1024, 0, stream>>>(tk_idx, tk_w, d_pos, d_w,
                                               slot_tok, tok_cnt);

  gather_kernel<<<NEXP * CAP, 128, 0, stream>>>(x, slot_tok, ein);

  // FFN1: [640x1024] @ [1024x4096] + b1, exact GELU
  expert_gemm_bf16_kernel<<<dim3(DHID / BN, CAP / BM, NEXP), 256, 0, stream>>>(
      ein, W1, b1, hbuf, CAP, DHID, DMODEL, 1);
  // FFN2: [640x4096] @ [4096x1024] + b2
  expert_gemm_bf16_kernel<<<dim3(DMODEL / BN, CAP / BM, NEXP), 256, 0, stream>>>(
      hbuf, W2, b2, eout, CAP, DMODEL, DHID, 0);

  combine_kernel<<<NTOK, 256, 0, stream>>>(eout, tk_idx, d_pos, d_w, y);
  aux_loss_kernel<<<1, 32, 0, stream>>>(colsum, tok_cnt,
                                        y + (size_t)NTOK * DMODEL);
}
